// SAE_76158360093019
// MI455X (gfx1250) — matrix operand order repaired and checked by reading, not compiled, not run on hardware
//
#include <hip/hip_runtime.h>
#include <hip/hip_bf16.h>

// ---------------------------------------------------------------------------
// SAE forward for MI455X (gfx1250, wave32, WMMA).
//   z   = relu(embs @ W_up^T + b_up)           (8192 x 24576)   -- WMMA bf16-split
//   kth = 32nd largest of each row of z
//   a   = z where z > kth else 0               (<=31 nnz/row)
//   out = a @ W_down^T                         (8192 x 768)     -- sparse gather
//
// v3: ROWS 32 -> 64 (four M-tiles per wave share one B register load) to halve
//     W_up L2 traffic again (~9.7 GB total). D_HID split in two halves per
//     row-group to keep 256 blocks of parallelism; per-block partial top-32
//     lists merged by a small rank-select kernel that also does the sparse
//     down-projection. Static LDS ~282 KB (CDNA5 WGP budget 320 KB).
// ---------------------------------------------------------------------------

typedef __attribute__((ext_vector_type(16))) __bf16 v16bf;
typedef __attribute__((ext_vector_type(8)))  float  v8f;

union BFV {
    uint4        q[2];
    unsigned int u[8];
    v16bf        v;
};

__device__ __forceinline__ unsigned short f32_to_bf16_rne(float x) {
    unsigned int u = __float_as_uint(x);
    unsigned int r = u + 0x7FFFu + ((u >> 16) & 1u);
    return (unsigned short)(r >> 16);
}
__device__ __forceinline__ float bf16_as_f32(unsigned short h) {
    return __uint_as_float(((unsigned int)h) << 16);
}

static constexpr int D_EMB = 768;
static constexpr int D_HID = 24576;
static constexpr int HHID  = D_HID / 2;  // 12288 columns per block (half)
static constexpr int B_SZ  = 8192;
static constexpr int ROWS  = 64;    // rows of embs per block (4 WMMA M tiles)
static constexpr int NTHR  = 512;   // 16 wave32
static constexpr int COLB  = 256;   // z columns per block iteration (16 waves x 16)
static constexpr int TOPK  = 32;
static constexpr int ZPAD  = 260;   // zstage row stride in dwords (260 % 64 = 4)
static constexpr int APAD  = 776;   // A row stride in ushorts (388 dw % 64 = 4)

// ---------------------------------------------------------------------------
// Kernel 0: W_down (768 x 24576) -> W_down^T (24576 x 768), tiled via LDS.
// ---------------------------------------------------------------------------
__global__ __launch_bounds__(256) void transpose_wdown(const float* __restrict__ Wd,
                                                       float* __restrict__ WdT) {
    __shared__ float tile[32][33];
    const int hb = blockIdx.x * 32;     // along D_HID
    const int eb = blockIdx.y * 32;     // along D_EMB
    const int tx = threadIdx.x;         // 0..31
    const int ty = threadIdx.y;         // 0..7
#pragma unroll
    for (int i = 0; i < 32; i += 8)
        tile[ty + i][tx] = Wd[(size_t)(eb + ty + i) * D_HID + hb + tx];
    __syncthreads();
#pragma unroll
    for (int i = 0; i < 32; i += 8)
        WdT[(size_t)(hb + ty + i) * D_EMB + eb + tx] = tile[tx][ty + i];
}

// ---------------------------------------------------------------------------
// Kernel 1: W_up fp32 -> bf16 hi/lo planes (split-precision operand prep).
// ---------------------------------------------------------------------------
__global__ __launch_bounds__(256) void convert_wup(const float* __restrict__ W,
                                                   unsigned short* __restrict__ hi,
                                                   unsigned short* __restrict__ lo,
                                                   int n) {
    for (int i = blockIdx.x * blockDim.x + threadIdx.x; i < n;
         i += gridDim.x * blockDim.x) {
        float x = W[i];
        unsigned short h = f32_to_bf16_rne(x);
        hi[i] = h;
        lo[i] = f32_to_bf16_rne(x - bf16_as_f32(h));
    }
}

// ---------------------------------------------------------------------------
// Kernel 2: fused up-GEMM (WMMA) + exact streaming top-32 over a column half.
// Grid: 256 blocks = 128 row-groups x 2 column halves. Block = 16 wave32; each
// wave owns FOUR 16x16 z tiles (M=0..63) per column-block, sharing one B
// register load (12 WMMAs per 64B of B per lane). Partial top-32 (val,idx)
// lists for this half go to workspace for the merge kernel.
// ---------------------------------------------------------------------------
__global__ __launch_bounds__(NTHR) void sae_up_topk(const float* __restrict__ embs,
                                                    const unsigned short* __restrict__ Whi,
                                                    const unsigned short* __restrict__ Wlo,
                                                    const float* __restrict__ b_up,
                                                    float* __restrict__ pv,
                                                    int* __restrict__ pi) {
    __shared__ __align__(16) unsigned short a_hi[ROWS][APAD];   // 97 KB
    __shared__ __align__(16) unsigned short a_lo[ROWS][APAD];   // 97 KB
    __shared__ float zstage[ROWS][ZPAD];                        // 66.6 KB
    __shared__ float topv[ROWS][33];                            //  8.4 KB
    __shared__ int   topi[ROWS][33];                            //  8.4 KB

    const int tid    = threadIdx.x;
    const int wave   = tid >> 5;    // 0..15
    const int lane   = tid & 31;
    const int half   = lane >> 4;   // 0: lanes 0-15, 1: lanes 16-31
    const int l16    = lane & 15;
    const int rowg0  = (blockIdx.x >> 1) * ROWS;   // row group
    const int chalf  = blockIdx.x & 1;             // which 12288-column half
    const int cstart = chalf * HHID;

    // ---- Stage A (64 x 768) into LDS as bf16 hi/lo split -------------------
    for (int i = tid; i < ROWS * D_EMB; i += NTHR) {
        int r = i / D_EMB, k = i - r * D_EMB;
        float x = embs[(size_t)(rowg0 + r) * D_EMB + k];
        unsigned short h = f32_to_bf16_rne(x);
        a_hi[r][k] = h;
        a_lo[r][k] = f32_to_bf16_rne(x - bf16_as_f32(h));
    }
    if (tid < ROWS) {
#pragma unroll
        for (int j = 0; j < TOPK; ++j) { topv[tid][j] = -1e30f; topi[tid][j] = 0; }
    }
    __syncthreads();

    float minv = -1e30f;   // running min of this thread's row top-32 (tid<ROWS)
    int   minpos = 0;

    // ---- Main loop over this half's 12288 z columns in blocks of 256 -------
    for (int cb = cstart; cb < cstart + HHID; cb += COLB) {
        const int colg = cb + wave * 16 + l16;           // z column = W_up row
        const unsigned short* bh = Whi + (size_t)colg * D_EMB;
        const unsigned short* bl = Wlo + (size_t)colg * D_EMB;

        v8f c0 = {}, c1 = {}, c2 = {}, c3 = {};          // M tiles 0..3
        for (int kb = 0; kb < D_EMB; kb += 32) {
            // A operand (ISA 7.12.2, 16-bit A 16x32): lane m = row, VGPR0-3 hold
            // K = 8*half.., VGPR4-7 hold K+16 -> dwords p0..p0+3 and p0+8..p0+11.
            const int p0 = (kb >> 1) + 4 * half;
            BFV Ah0, Al0, Ah1, Al1, Ah2, Al2, Ah3, Al3, Bh, Bl;
            Ah0.q[0] = ((const uint4*)&a_hi[l16][2 * p0])[0];
            Ah0.q[1] = ((const uint4*)&a_hi[l16][2 * p0])[2];
            Al0.q[0] = ((const uint4*)&a_lo[l16][2 * p0])[0];
            Al0.q[1] = ((const uint4*)&a_lo[l16][2 * p0])[2];
            Ah1.q[0] = ((const uint4*)&a_hi[16 + l16][2 * p0])[0];
            Ah1.q[1] = ((const uint4*)&a_hi[16 + l16][2 * p0])[2];
            Al1.q[0] = ((const uint4*)&a_lo[16 + l16][2 * p0])[0];
            Al1.q[1] = ((const uint4*)&a_lo[16 + l16][2 * p0])[2];
            Ah2.q[0] = ((const uint4*)&a_hi[32 + l16][2 * p0])[0];
            Ah2.q[1] = ((const uint4*)&a_hi[32 + l16][2 * p0])[2];
            Al2.q[0] = ((const uint4*)&a_lo[32 + l16][2 * p0])[0];
            Al2.q[1] = ((const uint4*)&a_lo[32 + l16][2 * p0])[2];
            Ah3.q[0] = ((const uint4*)&a_hi[48 + l16][2 * p0])[0];
            Ah3.q[1] = ((const uint4*)&a_hi[48 + l16][2 * p0])[2];
            Al3.q[0] = ((const uint4*)&a_lo[48 + l16][2 * p0])[0];
            Al3.q[1] = ((const uint4*)&a_lo[48 + l16][2 * p0])[2];
            // B operand (32x16 bf16): lane n = column; same K order as A:
            // K = 8*half..+7 and K = 16+8*half..+7 (two 16B chunks).
            Bh.q[0] = ((const uint4*)(bh + kb + 8 * half))[0];
            Bh.q[1] = ((const uint4*)(bh + kb + 8 * half))[2];
            Bl.q[0] = ((const uint4*)(bl + kb + 8 * half))[0];
            Bl.q[1] = ((const uint4*)(bl + kb + 8 * half))[2];

            c0 = __builtin_amdgcn_wmma_f32_16x16x32_bf16(false, Ah0.v, false, Bh.v,
                                                         (short)0, c0, false, false);
            c1 = __builtin_amdgcn_wmma_f32_16x16x32_bf16(false, Ah1.v, false, Bh.v,
                                                         (short)0, c1, false, false);
            c2 = __builtin_amdgcn_wmma_f32_16x16x32_bf16(false, Ah2.v, false, Bh.v,
                                                         (short)0, c2, false, false);
            c3 = __builtin_amdgcn_wmma_f32_16x16x32_bf16(false, Ah3.v, false, Bh.v,
                                                         (short)0, c3, false, false);
            c0 = __builtin_amdgcn_wmma_f32_16x16x32_bf16(false, Ah0.v, false, Bl.v,
                                                         (short)0, c0, false, false);
            c1 = __builtin_amdgcn_wmma_f32_16x16x32_bf16(false, Ah1.v, false, Bl.v,
                                                         (short)0, c1, false, false);
            c2 = __builtin_amdgcn_wmma_f32_16x16x32_bf16(false, Ah2.v, false, Bl.v,
                                                         (short)0, c2, false, false);
            c3 = __builtin_amdgcn_wmma_f32_16x16x32_bf16(false, Ah3.v, false, Bl.v,
                                                         (short)0, c3, false, false);
            c0 = __builtin_amdgcn_wmma_f32_16x16x32_bf16(false, Al0.v, false, Bh.v,
                                                         (short)0, c0, false, false);
            c1 = __builtin_amdgcn_wmma_f32_16x16x32_bf16(false, Al1.v, false, Bh.v,
                                                         (short)0, c1, false, false);
            c2 = __builtin_amdgcn_wmma_f32_16x16x32_bf16(false, Al2.v, false, Bh.v,
                                                         (short)0, c2, false, false);
            c3 = __builtin_amdgcn_wmma_f32_16x16x32_bf16(false, Al3.v, false, Bh.v,
                                                         (short)0, c3, false, false);
        }

        const float bias = b_up[colg];

        __syncthreads();                 // previous iteration's top-k scan done
        // C layout: VGPR r -> M = r + 8*half, N = l16.
        const int col = wave * 16 + l16;
#pragma unroll
        for (int r = 0; r < 8; ++r) {
            float v0 = c0[r] + bias; v0 = v0 > 0.f ? v0 : 0.f;
            zstage[r + 8 * half][col] = v0;
            float v1 = c1[r] + bias; v1 = v1 > 0.f ? v1 : 0.f;
            zstage[16 + r + 8 * half][col] = v1;
            float v2 = c2[r] + bias; v2 = v2 > 0.f ? v2 : 0.f;
            zstage[32 + r + 8 * half][col] = v2;
            float v3 = c3[r] + bias; v3 = v3 > 0.f ? v3 : 0.f;
            zstage[48 + r + 8 * half][col] = v3;
        }
        __syncthreads();

        // Exact streaming top-32: one thread per row scans 256 candidates.
        if (tid < ROWS) {
            float mv = minv; int mp = minpos;
            for (int j = 0; j < COLB; ++j) {
                float v = zstage[tid][j];
                if (v > mv) {
                    topv[tid][mp] = v;
                    topi[tid][mp] = cb + j;       // global column index
                    mv = topv[tid][0]; mp = 0;
#pragma unroll
                    for (int t = 1; t < TOPK; ++t) {
                        float tv = topv[tid][t];
                        if (tv < mv) { mv = tv; mp = t; }
                    }
                }
            }
            minv = mv; minpos = mp;
        }
    }

    __syncthreads();
    // ---- Emit this half's partial top-32 lists ----------------------------
    // Layout: candidate c of row R: pv[R*64 + chalf*32 + c].
    for (int i = tid; i < ROWS * TOPK; i += NTHR) {
        int r = i >> 5, j = i & 31;
        size_t o = (size_t)(rowg0 + r) * (2 * TOPK) + chalf * TOPK + j;
        pv[o] = topv[r][j];
        pi[o] = topi[r][j];
    }
}

// ---------------------------------------------------------------------------
// Kernel 3: merge the two partial top-32 lists (union contains the global
// top-32), find kth via rank selection, sparse down-projection.
// One block per batch row; candidates in LDS; gathers are coalesced rows of
// the L2-resident W_down^T.
// ---------------------------------------------------------------------------
__global__ __launch_bounds__(256) void merge_gather(const float* __restrict__ pv,
                                                    const int* __restrict__ pi,
                                                    const float* __restrict__ WdT,
                                                    float* __restrict__ out) {
    __shared__ float cv[2 * TOPK];
    __shared__ int   ci[2 * TOPK];
    __shared__ int   kbits;

    const int row = blockIdx.x;
    const int tid = threadIdx.x;

    if (tid < 2 * TOPK) {
        cv[tid] = pv[(size_t)row * (2 * TOPK) + tid];
        ci[tid] = pi[(size_t)row * (2 * TOPK) + tid];
    }
    if (tid == 0) kbits = 0;
    __syncthreads();

    // kth = 32nd largest of the 64 candidates = max of values with >=31
    // strictly-greater candidates. Values are post-ReLU (>=0), so float bits
    // compare monotonically as ints.
    if (tid < 2 * TOPK) {
        float v = cv[tid];
        int rank = 0;
#pragma unroll
        for (int j = 0; j < 2 * TOPK; ++j) rank += (cv[j] > v) ? 1 : 0;
        if (rank >= TOPK - 1) atomicMax(&kbits, __float_as_int(v));
    }
    __syncthreads();
    const float kth = __int_as_float(kbits);

    // out[row,:] = sum_{v>kth} v * WdT[idx,:]  (strict '>' excludes kth/ties)
    for (int e = tid; e < D_EMB; e += 256) {
        float acc = 0.f;
#pragma unroll 8
        for (int j = 0; j < 2 * TOPK; ++j) {
            float v = cv[j];
            if (v > kth)
                acc += v * WdT[(size_t)ci[j] * D_EMB + e];
        }
        out[(size_t)row * D_EMB + e] = acc;
    }
}

// ---------------------------------------------------------------------------
// Host launcher. Workspace layout (155.2 MB):
//   [0,          75497472)   : W_down^T  (24576 x 768 fp32)
//   [75497472,  113246208)   : W_up hi   (bf16)
//   [113246208, 150994944)   : W_up lo   (bf16)
//   [150994944, 153092096)   : partial top-k values (8192 x 64 fp32)
//   [153092096, 155189248)   : partial top-k indices (8192 x 64 int32)
// ---------------------------------------------------------------------------
extern "C" void kernel_launch(void* const* d_in, const int* in_sizes, int n_in,
                              void* d_out, int out_size, void* d_ws, size_t ws_size,
                              hipStream_t stream) {
    const float* embs   = (const float*)d_in[0];
    const float* W_up   = (const float*)d_in[1];
    const float* b_up   = (const float*)d_in[2];
    const float* W_down = (const float*)d_in[3];
    float*       outp   = (float*)d_out;

    char* ws = (char*)d_ws;
    float*          WdT = (float*)ws;
    unsigned short* Whi = (unsigned short*)(ws + (size_t)D_HID * D_EMB * 4);
    unsigned short* Wlo = Whi + (size_t)D_HID * D_EMB;
    float*          pv  = (float*)(ws + (size_t)D_HID * D_EMB * 8);
    int*            pi  = (int*)(ws + (size_t)D_HID * D_EMB * 8 +
                                 (size_t)B_SZ * 2 * TOPK * 4);

    // Phase 0: transpose W_down for coalesced sparse gathers (L2-resident).
    transpose_wdown<<<dim3(D_HID / 32, D_EMB / 32), dim3(32, 8), 0, stream>>>(W_down, WdT);

    // Phase 1: split W_up into bf16 hi/lo planes.
    convert_wup<<<2048, 256, 0, stream>>>(W_up, Whi, Wlo, D_HID * D_EMB);

    // Phase 2: WMMA up-projection + per-half top-32. 128 row-groups x 2 halves.
    sae_up_topk<<<(B_SZ / ROWS) * 2, NTHR, 0, stream>>>(embs, Whi, Wlo, b_up, pv, pi);

    // Phase 3: merge halves, rank-select kth, sparse down-projection.
    merge_gather<<<B_SZ, 256, 0, stream>>>(pv, pi, WdT, outp);
}
